// WModel_74363063763136
// MI455X (gfx1250) — matrix operand order repaired and checked by reading, not compiled, not run on hardware
//
#include <hip/hip_runtime.h>

typedef __bf16 bf16_t;
typedef __bf16 v2bf  __attribute__((ext_vector_type(2)));
typedef __bf16 v16bf __attribute__((ext_vector_type(16)));
typedef float  v8f   __attribute__((ext_vector_type(8)));

#define TLEN 4096
#define NB   32
#define LDSP 80      // padded LDS row pitch (bf16 elems) to cut bank conflicts
#define CF_LRELU 1

// ---------------- DWT / IDWT ----------------
__global__ void dwt_kernel(const float* __restrict__ x, float* __restrict__ out) {
  int t = blockIdx.x * 256 + threadIdx.x;
  int n = blockIdx.y;
  if (t >= TLEN) return;
  const float inv = 0.70710678118654752440f;
  float e = x[(size_t)n * 8192 + 2 * t];
  float o = x[(size_t)n * 8192 + 2 * t + 1];
  out[((size_t)n * 2 + 0) * TLEN + t] = (e + o) * inv;
  out[((size_t)n * 2 + 1) * TLEN + t] = (e - o) * inv;
}

__global__ void idwt_kernel(const float* __restrict__ o2, float* __restrict__ y) {
  int t = blockIdx.x * 256 + threadIdx.x;
  int n = blockIdx.y;
  if (t >= TLEN) return;
  const float inv = 0.70710678118654752440f;
  float L = o2[((size_t)n * 2 + 0) * TLEN + t];
  float H = o2[((size_t)n * 2 + 1) * TLEN + t];
  y[(size_t)n * 8192 + 2 * t]     = (L + H) * inv;
  y[(size_t)n * 8192 + 2 * t + 1] = (L - H) * inv;
}

// ------------- positional encoding + linear (per projection) -------------
__global__ void emb_kernel(const float* __restrict__ nl, const float* __restrict__ lw,
                           const float* __restrict__ lb, float* __restrict__ embp) {
  int n = blockIdx.x, co = threadIdx.x;  // blockDim = 64
  float s = nl[n];
  float acc = lb[co];
  for (int i = 0; i < 64; ++i) {
    int ii = i & 31;
    float enc = s * __expf(-0.28782313662425572f * (float)ii);  // ln(1e4)/32
    float e = (i < 32) ? __sinf(enc) : __cosf(enc);
    acc += e * lw[co * 64 + i];
  }
  embp[(n << 6) + co] = acc;
}

// ------------- pack f32 conv weights into WMMA A-fragment bf16 layout -------------
// K ordering: chunk c covers tap j = c/2, input channels [32*(c&1), +32).
// Within a chunk: 16x32 bf16 A layout (lanes 0-15: K 0..15;
// lanes 16-31: K 16..31; M = lane%16).
__global__ void pack_w_kernel(const float* __restrict__ w, bf16_t* __restrict__ pw,
                              int CO, int K) {
  int idx = blockIdx.x * 256 + threadIdx.x;
  int total = CO * 64 * K;
  if (idx >= total) return;
  int e = idx & 15;
  int r = idx >> 4;
  int lane = r & 31; r >>= 5;
  int chunk = r % (2 * K);
  int cotile = r / (2 * K);
  int j = chunk >> 1, half = chunk & 1;
  int M = lane & 15, hi = lane >> 4;
  int v = e >> 1, pos = e & 1;
  int kk = hi * 16 + e;
  int ci = half * 32 + kk;
  int co = cotile * 16 + M;
  pw[idx] = (bf16_t)w[((size_t)co * 64 + ci) * K + j];
}

// ------------- main implicit-GEMM conv (Cin=64) via V_WMMA_F32_16X16X32_BF16 -------------
// Block = 128 threads (4 waves), block tile = 256 time positions.
// Each wave owns its private 64-wide time span and computes ALL NCO co-tiles,
// so every B fragment (LDS) feeds NCO WMMAs and every A fragment feeds 4 WMMAs.
// K and DIL are template parameters -> fully unrolled chunk loop, constexpr tile
// geometry, and all LDS B-fragment addresses fold into DS immediate offsets.
template <int K, int DIL, int NCO>
__global__ __launch_bounds__(128) void conv_wmma_kernel(
    const float* __restrict__ in,    // (N,64,T)
    float* __restrict__ out,         // (N,64,T), write at channel offset co_off
    const bf16_t* __restrict__ pw,   // packed A fragments
    const float* __restrict__ bias,  // (CO)
    const float* __restrict__ embp,  // (N,64) or null
    const float* __restrict__ add0,  // (N,64,T) or null
    const float* __restrict__ add1,  // (N,64,T) or null
    int co_off, int flags) {
  constexpr int PAD = DIL * (K - 1) / 2;
  constexpr int TW = 256 + (K - 1) * DIL;   // <= 288
  int lane = threadIdx.x & 31;
  int wave = threadIdx.x >> 5;
  int n = blockIdx.y;
  int t0 = blockIdx.x << 8;

  __shared__ alignas(32) bf16_t lds[TW * LDSP];

  // stage input tile (time-major rows, channel-contiguous) with reflect pad.
  // wave -> channel-pair (strided), lane -> time (strided by 32): no div/mod,
  // global loads coalesced along t, two channels packed per b32 DS store.
  for (int cp = wave; cp < 32; cp += 4) {
    int ci = cp * 2;
    const float* src = in + ((size_t)(n << 6) + ci) * TLEN;
    for (int tt = lane; tt < TW; tt += 32) {
      int s = t0 - PAD + tt;
      s = (s < 0) ? -s : s;
      s = (s >= TLEN) ? (2 * TLEN - 2 - s) : s;
      v2bf p;
      p.x = (bf16_t)src[s];
      p.y = (bf16_t)src[TLEN + s];
      *(v2bf*)(lds + tt * LDSP + ci) = p;
    }
  }
  __syncthreads();

  int khalf = lane >> 4, ncol = lane & 15;
  int twave = wave << 6;  // this wave's time offset within the block tile
  const bf16_t* ldsbase = lds + (twave + ncol) * LDSP + khalf * 16;
  v8f acc[NCO][4] = {};

#pragma unroll
  for (int j = 0; j < K; ++j) {
#pragma unroll
    for (int half = 0; half < 2; ++half) {
      const int chunk = j * 2 + half;
      v16bf a[NCO];
#pragma unroll
      for (int c = 0; c < NCO; ++c)
        a[c] = *(const v16bf*)(pw + (((size_t)c * (2 * K) + chunk) * 32 + lane) * 16);
#pragma unroll
      for (int nt = 0; nt < 4; ++nt) {
        // constexpr offset from per-thread base -> DS immediate offsets
        const v16bf b = *(const v16bf*)(ldsbase + (nt * 16 + j * DIL) * LDSP + half * 32);
#pragma unroll
        for (int c = 0; c < NCO; ++c)
          acc[c][nt] = __builtin_amdgcn_wmma_f32_16x16x32_bf16(
              false, a[c], false, b, (short)0, acc[c][nt], false, false);
      }
    }
  }

  // epilogue: bias (+emb) (+lrelu) (+add0) (+add1)
#pragma unroll
  for (int c = 0; c < NCO; ++c) {
#pragma unroll
    for (int nt = 0; nt < 4; ++nt) {
      int t = t0 + twave + nt * 16 + ncol;
#pragma unroll
      for (int r = 0; r < 8; ++r) {
        int co = (c << 4) + r + (khalf << 3);
        float y = acc[c][nt][r] + bias[co];
        if (embp) y += embp[(n << 6) + co];
        if (flags & CF_LRELU) y = (y > 0.f) ? y : 0.2f * y;
        size_t o = ((size_t)(n << 6) + co_off + co) * TLEN + t;
        if (add0) y += add0[o];
        if (add1) y += add1[o];
        out[o] = y;
      }
    }
  }
}

// ------------- conv0: Cin=2, k=9, lrelu (+optional add) -------------
__global__ void conv0_kernel(const float* __restrict__ in, float* __restrict__ out,
                             const float* __restrict__ w, const float* __restrict__ b,
                             const float* __restrict__ add) {
  int t = blockIdx.x * 256 + threadIdx.x;
  int co = blockIdx.y, n = blockIdx.z;
  if (t >= TLEN) return;
  float y = b[co];
  for (int ci = 0; ci < 2; ++ci)
    for (int j = 0; j < 9; ++j) {
      int s = t - 4 + j;
      s = (s < 0) ? -s : s;
      s = (s >= TLEN) ? (2 * TLEN - 2 - s) : s;
      y += w[(co * 2 + ci) * 9 + j] * in[((size_t)n * 2 + ci) * TLEN + s];
    }
  y = (y > 0.f) ? y : 0.2f * y;
  size_t o = ((size_t)(n << 6) + co) * TLEN + t;
  if (add) y += add[o];
  out[o] = y;
}

// ------------- conv_out: Cin=64, Cout=2, k=9 -------------
__global__ void convout_kernel(const float* __restrict__ in, float* __restrict__ out2,
                               const float* __restrict__ w, const float* __restrict__ b) {
  int t = blockIdx.x * 256 + threadIdx.x;
  int co = blockIdx.y, n = blockIdx.z;
  if (t >= TLEN) return;
  float y = b[co];
  for (int ci = 0; ci < 64; ++ci)
    for (int j = 0; j < 9; ++j) {
      int s = t - 4 + j;
      s = (s < 0) ? -s : s;
      s = (s >= TLEN) ? (2 * TLEN - 2 - s) : s;
      y += w[(co * 64 + ci) * 9 + j] * in[((size_t)(n << 6) + ci) * TLEN + s];
    }
  out2[((size_t)n * 2 + co) * TLEN + t] = y;
}

// ------------- instance-norm (ch<32) + lrelu (all 64 ch), in place -------------
__global__ __launch_bounds__(256) void instnorm_lrelu_kernel(float* __restrict__ h) {
  int c = blockIdx.x & 63;
  int n = blockIdx.x >> 6;
  float* row = h + ((size_t)(n << 6) + c) * TLEN;
  if (c < 32) {
    __shared__ float sm[256], sv[256];
    float m = 0.f, v = 0.f;
    for (int t = threadIdx.x; t < TLEN; t += 256) {
      float x = row[t];
      m += x; v += x * x;
    }
    sm[threadIdx.x] = m; sv[threadIdx.x] = v;
    __syncthreads();
    for (int s = 128; s > 0; s >>= 1) {
      if (threadIdx.x < s) { sm[threadIdx.x] += sm[threadIdx.x + s]; sv[threadIdx.x] += sv[threadIdx.x + s]; }
      __syncthreads();
    }
    float mean = sm[0] * (1.f / TLEN);
    float var = sv[0] * (1.f / TLEN) - mean * mean;
    float inv = rsqrtf(var + 1e-5f);
    for (int t = threadIdx.x; t < TLEN; t += 256) {
      float y = (row[t] - mean) * inv;
      row[t] = (y > 0.f) ? y : 0.2f * y;
    }
  } else {
    for (int t = threadIdx.x; t < TLEN; t += 256) {
      float y = row[t];
      row[t] = (y > 0.f) ? y : 0.2f * y;
    }
  }
}

// =================== host orchestration ===================
extern "C" void kernel_launch(void* const* d_in, const int* in_sizes, int n_in,
                              void* d_out, int out_size, void* d_ws, size_t ws_size,
                              hipStream_t stream) {
  auto F32 = [&](int i) { return (const float*)d_in[i]; };

  // Input indices assuming jax pytree flatten (sorted dict keys) of setup_inputs():
  // 0:x 1:cond 2:noise_scale; params: conv_out{b,w}; proj[p]{in_conv{b,w},lin_b,lin_w,out_conv{b,w}};
  // stream_cond{conv0{b,w}, rb[i]{conv1{b,w},conv2{b,w},filters[f]{b,w}}}; stream_x{...}
  const int IDX_COND = 1, IDX_NS = 2;
  const int IDX_CONVOUT_B = 3, IDX_CONVOUT_W = 4;
  const int PROJ0 = 5;   // + p*6 : in_b, in_w, lin_b, lin_w, out_b, out_w
  const int SC = 29, SX = 67;  // conv0_b, conv0_w; rb[i] at +2+i*12

  char* ws = (char*)d_ws;
  const size_t ACT = (size_t)NB * 64 * TLEN * sizeof(float);
  float* XS[4];
  for (int i = 0; i < 4; ++i) XS[i] = (float*)(ws + i * ACT);
  float* H  = (float*)(ws + 4 * ACT);
  float* Fb = (float*)(ws + 5 * ACT);
  float* T1 = (float*)(ws + 6 * ACT);
  size_t off = 7 * ACT;
  float* DWTX = (float*)(ws + off); off += (size_t)NB * 2 * TLEN * 4;
  float* DWTC = (float*)(ws + off); off += (size_t)NB * 2 * TLEN * 4;
  float* OUT2 = (float*)(ws + off); off += (size_t)NB * 2 * TLEN * 4;
  float* EMBP = (float*)(ws + off); off += (size_t)4 * NB * 64 * 4;
  bf16_t* pwbase = (bf16_t*)(ws + off);
  size_t pwcur = 0;

  auto pack = [&](int widx, int CO, int K) -> const bf16_t* {
    bf16_t* dst = pwbase + pwcur;
    int total = CO * 64 * K;
    pwcur += (size_t)total;
    pack_w_kernel<<<dim3((total + 255) / 256), dim3(256), 0, stream>>>(F32(widx), dst, CO, K);
    return dst;
  };

  auto conv = [&](const float* in, float* outp, const bf16_t* pw, int bidx,
                  int K, int dil, const float* emb, const float* a0, const float* a1,
                  int co_off, int nco, int flags) {
    dim3 g(TLEN / 256, NB), b(128);
    const float* bs = F32(bidx);
#define LCH(KK, DD, NN) \
    conv_wmma_kernel<KK, DD, NN><<<g, b, 0, stream>>>(in, outp, pw, bs, emb, a0, a1, co_off, flags)
    if (nco == 4) {
      if (K == 3) LCH(3, 1, 4); else LCH(9, 1, 4);
    } else if (dil == 1) {
      switch (K) { case 3: LCH(3, 1, 1); break; case 5: LCH(5, 1, 1); break;
                   case 7: LCH(7, 1, 1); break; default: LCH(9, 1, 1); break; }
    } else if (dil == 2) {
      switch (K) { case 3: LCH(3, 2, 1); break; case 5: LCH(5, 2, 1); break;
                   case 7: LCH(7, 2, 1); break; default: LCH(9, 2, 1); break; }
    } else {
      switch (K) { case 3: LCH(3, 4, 1); break; case 5: LCH(5, 4, 1); break;
                   case 7: LCH(7, 4, 1); break; default: LCH(9, 4, 1); break; }
    }
#undef LCH
  };

  // pack all WMMA-path weights (bf16 A-fragments)
  const bf16_t *pwProjIn[4], *pwProjOut[4];
  for (int p = 0; p < 4; ++p) {
    pwProjIn[p]  = pack(PROJ0 + p * 6 + 1, 64, 3);
    pwProjOut[p] = pack(PROJ0 + p * 6 + 5, 64, 3);
  }
  const bf16_t *pwC1[2][3], *pwC2[2][3], *pwFl[2][3][4];
  const int SB[2] = {SX, SC};
  for (int s = 0; s < 2; ++s)
    for (int i = 0; i < 3; ++i) {
      int rb = SB[s] + 2 + i * 12;
      pwC1[s][i] = pack(rb + 1, 64, 9);
      pwC2[s][i] = pack(rb + 3, 64, 9);
      for (int f = 0; f < 4; ++f)
        pwFl[s][i][f] = pack(rb + 4 + 2 * f + 1, 16, 3 + 2 * f);
    }

  // embeddings per projection: emb @ lin_w.T + lin_b -> (N,64)
  for (int p = 0; p < 4; ++p)
    emb_kernel<<<dim3(NB), dim3(64), 0, stream>>>(
        F32(IDX_NS), F32(PROJ0 + p * 6 + 3), F32(PROJ0 + p * 6 + 2), EMBP + (size_t)p * NB * 64);

  const int DIL[3] = {1, 2, 4};

  // ---------------- stream_x ----------------
  dwt_kernel<<<dim3(16, NB), dim3(256), 0, stream>>>(F32(0), DWTX);
  conv0_kernel<<<dim3(16, 64, NB), dim3(256), 0, stream>>>(DWTX, H, F32(SX + 1), F32(SX + 0), nullptr);
  // projection 0
  conv(H, T1, pwProjIn[0], PROJ0 + 0, 3, 1, EMBP, nullptr, nullptr, 0, 4, 0);
  conv(T1, XS[0], pwProjOut[0], PROJ0 + 4, 3, 1, nullptr, nullptr, nullptr, 0, 4, 0);
  for (int i = 0; i < 3; ++i) {
    int rb = SX + 2 + i * 12, d = DIL[i];
    for (int f = 0; f < 4; ++f)
      conv(H, Fb, pwFl[0][i][f], rb + 4 + 2 * f, 3 + 2 * f, d, nullptr, nullptr, nullptr, f * 16, 1, 0);
    conv(Fb, T1, pwC1[0][i], rb + 0, 9, 1, nullptr, nullptr, nullptr, 0, 4, 0);
    instnorm_lrelu_kernel<<<dim3(NB * 64), dim3(256), 0, stream>>>(T1);
    conv(T1, H, pwC2[0][i], rb + 2, 9, 1, nullptr, H, nullptr, 0, 4, CF_LRELU);
    int p = i + 1;
    conv(H, T1, pwProjIn[p], PROJ0 + p * 6 + 0, 3, 1, EMBP + (size_t)p * NB * 64, nullptr, nullptr, 0, 4, 0);
    conv(T1, XS[p], pwProjOut[p], PROJ0 + p * 6 + 4, 3, 1, nullptr, nullptr, nullptr, 0, 4, 0);
  }

  // ---------------- stream_cond ----------------
  dwt_kernel<<<dim3(16, NB), dim3(256), 0, stream>>>(F32(IDX_COND), DWTC);
  conv0_kernel<<<dim3(16, 64, NB), dim3(256), 0, stream>>>(DWTC, H, F32(SC + 1), F32(SC + 0), XS[0]);
  for (int i = 0; i < 3; ++i) {
    int rb = SC + 2 + i * 12, d = DIL[i];
    for (int f = 0; f < 4; ++f)
      conv(H, Fb, pwFl[1][i][f], rb + 4 + 2 * f, 3 + 2 * f, d, nullptr, nullptr, nullptr, f * 16, 1, 0);
    conv(Fb, T1, pwC1[1][i], rb + 0, 9, 1, nullptr, nullptr, nullptr, 0, 4, 0);
    instnorm_lrelu_kernel<<<dim3(NB * 64), dim3(256), 0, stream>>>(T1);
    conv(T1, H, pwC2[1][i], rb + 2, 9, 1, nullptr, H, XS[i + 1], 0, 4, CF_LRELU);
  }
  convout_kernel<<<dim3(16, 2, NB), dim3(256), 0, stream>>>(H, OUT2, F32(IDX_CONVOUT_W), F32(IDX_CONVOUT_B));
  idwt_kernel<<<dim3(16, NB), dim3(256), 0, stream>>>(OUT2, (float*)d_out);

  (void)in_sizes; (void)n_in; (void)out_size; (void)ws_size;
}
